// SkipHashingMemory_34763465294233
// MI455X (gfx1250) — matrix operand order repaired and checked by reading, not compiled, not run on hardware
//
#include <hip/hip_runtime.h>
#include <hip/hip_bf16.h>

// ---------------- problem constants ----------------
#define NB    8
#define TT    1024
#define NTOK  (NB * TT)          // 8192
#define DIM   1024
#define HEADS 4
#define KDIM  256
#define HALF  128
#define NKEYS 1024
#define KNN   32
#define VDIM  32

typedef __attribute__((ext_vector_type(16))) _Float16 v16h;
typedef __attribute__((ext_vector_type(8)))  _Float16 v8h;
typedef __attribute__((ext_vector_type(8)))  float    v8f;

// ---------------- WMMA helpers (gfx1250, wave32) ----------------
__device__ __forceinline__ v8f wmma_f16(v16h a, v16h b, v8f c) {
  // 8 args: (neg_a, A, neg_b, B, c_mod, C, reuse_a, reuse_b)
  return __builtin_amdgcn_wmma_f32_16x16x32_f16(false, a, false, b, (short)0, c,
                                                false, false);
}

// A fragment, 16x32 f16, from row-major [M][K] matrix.
// lane<16 holds row m=lane, K in {8s..8s+7} u {16+8s..16+8s+7} with s=lane>>4.
__device__ __forceinline__ v16h load_frag_a(const _Float16* base, int ld,
                                            int row0, int k0, int lane) {
  int m = lane & 15, sel = lane >> 4;
  const _Float16* p = base + (size_t)(row0 + m) * ld + k0 + 8 * sel;
  v8h lo = *(const v8h*)p;
  v8h hi = *(const v8h*)(p + 16);
  v16h a;
#pragma unroll
  for (int t = 0; t < 8; ++t) { a[t] = lo[t]; a[t + 8] = hi[t]; }
  return a;
}

// B fragment, 32x16 f16, from TRANSPOSED row-major [N][K] matrix.
// lane holds column n=lane&15, K in {8s..8s+7} u {16+8s..16+8s+7} with s=lane>>4.
__device__ __forceinline__ v16h load_frag_b(const _Float16* baseT, int ld,
                                            int col0, int k0, int lane) {
  int n = lane & 15, sel = lane >> 4;
  const _Float16* p = baseT + (size_t)(col0 + n) * ld + k0 + 8 * sel;
  v8h lo = *(const v8h*)p;
  v8h hi = *(const v8h*)(p + 16);
  v16h b;
#pragma unroll
  for (int t = 0; t < 8; ++t) { b[t] = lo[t]; b[t + 8] = hi[t]; }
  return b;
}

// ---------------- conversion kernels ----------------
__global__ void k_cvt_f16(const float* __restrict__ src,
                          _Float16* __restrict__ dst, int n) {
  int i = blockIdx.x * 256 + threadIdx.x;
  if (i < n) dst[i] = (_Float16)src[i];
}

// src[rows][cols] (f32) -> dst[cols][rows] (f16)
__global__ void k_transpose_f16(const float* __restrict__ src,
                                _Float16* __restrict__ dst, int rows, int cols) {
  int i = blockIdx.x * 256 + threadIdx.x;
  if (i < rows * cols) {
    int r = i / cols, c = i - r * cols;
    dst[(size_t)c * rows + r] = (_Float16)src[i];
  }
}

// ---------------- stage 1: q = f16(x @ W_q + b_q) ----------------
// grid (NTOK/16, DIM/64), block 128 (4 waves), wave -> one 16x16 col tile
__global__ __launch_bounds__(128) void k_qproj(const _Float16* __restrict__ xh,
                                               const _Float16* __restrict__ wqhT,
                                               const float* __restrict__ bq,
                                               _Float16* __restrict__ qh) {
  int wave = threadIdx.x >> 5, lane = threadIdx.x & 31;
  int row0 = blockIdx.x * 16;
  int col0 = blockIdx.y * 64 + wave * 16;
  v8f acc = {};
#pragma unroll 4
  for (int kt = 0; kt < 32; ++kt) {
    v16h a = load_frag_a(xh, DIM, row0, kt * 32, lane);
    v16h b = load_frag_b(wqhT, DIM, col0, kt * 32, lane);
    acc = wmma_f16(a, b, acc);
  }
  int m0 = (lane >> 4) * 8, n = lane & 15;
  float bias = bq[col0 + n];
#pragma unroll
  for (int r = 0; r < 8; ++r)
    qh[(size_t)(row0 + m0 + r) * (HEADS * KDIM) + col0 + n] =
        (_Float16)(acc[r] + bias);
}

// ---------------- stage 2: scores (WMMA) + per-row top-32 ----------------
// grid (NTOK/16, HEADS*2), block 256 (8 waves); dynamic LDS = 16*1024 floats
__global__ __launch_bounds__(256) void k_scores_topk(
    const _Float16* __restrict__ qh, const _Float16* __restrict__ keysh,
    float* __restrict__ sbuf, int* __restrict__ ibuf) {
  extern __shared__ float sc[];  // [16][NKEYS]
  int wave = threadIdx.x >> 5, lane = threadIdx.x & 31;
  int row0 = blockIdx.x * 16;
  int hc = blockIdx.y;  // h*2 + c
  const _Float16* qbase = qh + (size_t)hc * HALF;             // col offset in q
  const _Float16* kbase = keysh + (size_t)hc * NKEYS * HALF;  // [key][k] f16

  // Hoist the invariant 16x128 A tile (q) into registers: 4 fragments.
  v16h afrag[4];
#pragma unroll
  for (int kt = 0; kt < 4; ++kt)
    afrag[kt] = load_frag_a(qbase, HEADS * KDIM, row0, kt * 32, lane);

  for (int i = 0; i < 8; ++i) {
    int ct = wave * 8 + i;  // 16-key column tile
    if (i + 1 < 8)
      __builtin_prefetch(kbase + (size_t)(ct + 1) * 16 * HALF, 0, 1);
    v8f acc = {};
#pragma unroll
    for (int kt = 0; kt < 4; ++kt) {
      v16h b = load_frag_b(kbase, HALF, ct * 16, kt * 32, lane);
      acc = wmma_f16(afrag[kt], b, acc);
    }
    int m0 = (lane >> 4) * 8, n = lane & 15;
#pragma unroll
    for (int r = 0; r < 8; ++r) sc[(m0 + r) * NKEYS + ct * 16 + n] = acc[r];
  }
  __syncthreads();

  // iterative top-32: wave handles rows 2*wave, 2*wave+1 (descending order)
  for (int rr = 0; rr < 2; ++rr) {
    int m = wave * 2 + rr;
    float* row = &sc[m * NKEYS];
    size_t obase = (((size_t)(row0 + m)) * (HEADS * 2) + hc) * KNN;
    for (int r = 0; r < KNN; ++r) {
      float bv = -1e30f;
      int bi = 0;
#pragma unroll 8
      for (int j = 0; j < 32; ++j) {
        int idx = lane + (j << 5);
        float v = row[idx];
        if (v > bv) { bv = v; bi = idx; }
      }
#pragma unroll
      for (int off = 16; off; off >>= 1) {
        float ov = __shfl_xor(bv, off);
        int oi = __shfl_xor(bi, off);
        if (ov > bv) { bv = ov; bi = oi; }
      }
      row[bi] = -1e30f;  // all lanes agree on bi
      if (lane == 0) { sbuf[obase + r] = bv; ibuf[obase + r] = bi; }
    }
  }
}

// ---------------- stage 3: combine + top-32 + softmax + gather ----------------
// grid NTOK, block 128 (wave == head); lane == value dim for coalesced gathers
__global__ __launch_bounds__(128) void k_combine(const float* __restrict__ sbuf,
                                                 const int* __restrict__ ibuf,
                                                 const float* __restrict__ values,
                                                 _Float16* __restrict__ yh) {
  __shared__ float comb[HEADS][1024];
  __shared__ int i1sh[HEADS][KNN], i2sh[HEADS][KNN];
  __shared__ float sels[HEADS][KNN];
  __shared__ int seli[HEADS][KNN];
  __shared__ float ysh[HEADS][VDIM];

  int h = threadIdx.x >> 5, lane = threadIdx.x & 31;
  int n = blockIdx.x;
  size_t base = ((size_t)n * (HEADS * 2) + h * 2) * KNN;
  float s1 = sbuf[base + lane];
  float s2 = sbuf[base + KNN + lane];
  i1sh[h][lane] = ibuf[base + lane];
  i2sh[h][lane] = ibuf[base + KNN + lane];

  // comb[j*32 + l] = s1[j] + s2[l]
#pragma unroll 4
  for (int j = 0; j < 32; ++j) comb[h][(j << 5) + lane] = __shfl(s1, j) + s2;

  float* row = comb[h];
  for (int r = 0; r < KNN; ++r) {
    float bv = -1e30f;
    int bi = 0;
#pragma unroll 8
    for (int j = 0; j < 32; ++j) {
      int idx = lane + (j << 5);
      float v = row[idx];
      if (v > bv) { bv = v; bi = idx; }
    }
#pragma unroll
    for (int off = 16; off; off >>= 1) {
      float ov = __shfl_xor(bv, off);
      int oi = __shfl_xor(bi, off);
      if (ov > bv) { bv = ov; bi = oi; }
    }
    row[bi] = -1e30f;
    if (lane == 0) {
      sels[h][r] = bv;
      seli[h][r] = i1sh[h][bi >> 5] * NKEYS + i2sh[h][bi & 31];
    }
  }

  // softmax: selection was descending, so sels[h][0] is the max
  float sv = sels[h][lane];
  float w = __expf(sv - sels[h][0]);
  float sum = w;
#pragma unroll
  for (int off = 16; off; off >>= 1) sum += __shfl_xor(sum, off);
  w /= sum;
  int idx = seli[h][lane];

  // gather + weighted sum; lane = value dim -> one coalesced 128B row per k
  float acc = 0.f;
  for (int k = 0; k < KNN; ++k) {
    float wk = __shfl(w, k);
    int ik = __shfl(idx, k);
    acc += wk * values[(size_t)ik * VDIM + lane];
  }
  ysh[h][lane] = acc;
  __syncthreads();
  if (h == 0) {
    float y = ysh[0][lane] + ysh[1][lane] + ysh[2][lane] + ysh[3][lane];
    yh[(size_t)n * VDIM + lane] = (_Float16)y;
  }
}

// ---------------- stage 4: out = x + y @ W_r + b_r ----------------
// grid (NTOK/16, DIM/64), block 128; one K=32 WMMA per wave-tile
__global__ __launch_bounds__(128) void k_out(const _Float16* __restrict__ yh,
                                             const _Float16* __restrict__ wrhT,
                                             const float* __restrict__ x,
                                             const float* __restrict__ br,
                                             float* __restrict__ out) {
  int wave = threadIdx.x >> 5, lane = threadIdx.x & 31;
  int row0 = blockIdx.x * 16;
  int col0 = blockIdx.y * 64 + wave * 16;
  v16h a = load_frag_a(yh, VDIM, row0, 0, lane);
  v16h b = load_frag_b(wrhT, VDIM, col0, 0, lane);
  v8f acc = {};
  acc = wmma_f16(a, b, acc);
  int m0 = (lane >> 4) * 8, n = lane & 15;
  float bias = br[col0 + n];
#pragma unroll
  for (int r = 0; r < 8; ++r) {
    size_t o = (size_t)(row0 + m0 + r) * DIM + col0 + n;
    out[o] = x[o] + acc[r] + bias;
  }
}

// ---------------- workspace layout (bytes, 256-aligned) ----------------
#define OFF_XH    ((size_t)0)          // N*D f16      = 16,777,216
#define OFF_WQH   ((size_t)16777216)   // 1024*1024 f16 (transposed [out][in])
#define OFF_KEYSH ((size_t)18874368)   // H*2*NK*HALF f16 (already [key][k])
#define OFF_WRH   ((size_t)20971520)   // 1024*32 f16 (transposed [d][v])
#define OFF_QH    ((size_t)21037056)   // N*1024 f16
#define OFF_SBUF  ((size_t)37814272)   // N*H*2*KNN f32
#define OFF_IBUF  ((size_t)46202880)   // N*H*2*KNN i32
#define OFF_YH    ((size_t)54591488)   // N*32 f16

extern "C" void kernel_launch(void* const* d_in, const int* in_sizes, int n_in,
                              void* d_out, int out_size, void* d_ws,
                              size_t ws_size, hipStream_t stream) {
  const float* x      = (const float*)d_in[0];
  const float* W_q    = (const float*)d_in[1];
  const float* b_q    = (const float*)d_in[2];
  const float* keys   = (const float*)d_in[3];
  const float* values = (const float*)d_in[4];
  const float* W_r    = (const float*)d_in[5];
  const float* b_r    = (const float*)d_in[6];
  float* out = (float*)d_out;

  char* ws = (char*)d_ws;
  _Float16* xh    = (_Float16*)(ws + OFF_XH);
  _Float16* wqhT  = (_Float16*)(ws + OFF_WQH);
  _Float16* keysh = (_Float16*)(ws + OFF_KEYSH);
  _Float16* wrhT  = (_Float16*)(ws + OFF_WRH);
  _Float16* qh    = (_Float16*)(ws + OFF_QH);
  float*    sbuf  = (float*)(ws + OFF_SBUF);
  int*      ibuf  = (int*)(ws + OFF_IBUF);
  _Float16* yh    = (_Float16*)(ws + OFF_YH);

  // ---- conversions ----
  {
    int n = NTOK * DIM;  // x -> f16
    k_cvt_f16<<<(n + 255) / 256, 256, 0, stream>>>(x, xh, n);
  }
  {
    int n = HEADS * 2 * NKEYS * HALF;  // keys -> f16 (layout already [key][k])
    k_cvt_f16<<<(n + 255) / 256, 256, 0, stream>>>(keys, keysh, n);
  }
  {
    int n = DIM * (HEADS * KDIM);  // W_q [in][out] -> wqhT [out][in]
    k_transpose_f16<<<(n + 255) / 256, 256, 0, stream>>>(W_q, wqhT, DIM,
                                                         HEADS * KDIM);
  }
  {
    int n = VDIM * DIM;  // W_r [v][d] -> wrhT [d][v]
    k_transpose_f16<<<(n + 255) / 256, 256, 0, stream>>>(W_r, wrhT, VDIM, DIM);
  }

  // ---- stage 1: q projection ----
  k_qproj<<<dim3(NTOK / 16, DIM / 64), 128, 0, stream>>>(xh, wqhT, b_q, qh);

  // ---- stage 2: scores + first top-k (64KB dynamic LDS) ----
  k_scores_topk<<<dim3(NTOK / 16, HEADS * 2), 256, 16 * NKEYS * sizeof(float),
                  stream>>>(qh, keysh, sbuf, ibuf);

  // ---- stage 3: combine + final top-k + softmax + gather ----
  k_combine<<<NTOK, 128, 0, stream>>>(sbuf, ibuf, values, yh);

  // ---- stage 4: reprojection + residual ----
  k_out<<<dim3(NTOK / 16, DIM / 64), 128, 0, stream>>>(yh, wrhT, x, b_r, out);
}
